// LinearAttention_49830210568438
// MI455X (gfx1250) — matrix operand order repaired and checked by reading, not compiled, not run on hardware
//
#include <hip/hip_runtime.h>

typedef __bf16 bf16_t;
typedef __attribute__((ext_vector_type(16))) __bf16 v16bf;
typedef __attribute__((ext_vector_type(8)))  __bf16 v8bf;
typedef __attribute__((ext_vector_type(4)))  __bf16 v4bf;
typedef __attribute__((ext_vector_type(8)))  float  v8f;

#define N_TOT  65536      // 16*64*64
#define C_IN   256
#define TILE_N 128
#define PART_STRIDE (128 + 4096)   // S[128] + N[4][32][32]
#define NTILES 512                 // 65536 / 128

__device__ __forceinline__ v16bf cat16(v8bf lo, v8bf hi) {
  return __builtin_shufflevector(lo, hi, 0, 1, 2, 3, 4, 5, 6, 7,
                                         8, 9, 10, 11, 12, 13, 14, 15);
}

// ---- WMMA fragment helpers (layouts per CDNA5 ISA 7.12.2, wave32) ----
// A 16x32 bf16: lane<16: row m=lane, K {0..7,16..23}; lane>=16: row m=lane-16,
// K {8..15,24..31}.  Two contiguous 8-element (16B) runs per lane.
__device__ __forceinline__ v16bf frag_a(const bf16_t* __restrict__ p, int ld,
                                        int row0, int k0, int lane) {
  const int half = (lane >> 4) & 1;
  const int m = lane & 15;
  const bf16_t* base = p + (row0 + m) * ld + k0 + half * 8;
  const v8bf lo = *(const v8bf*)(base);       // e=0..7  : k = k0+half*8+e
  const v8bf hi = *(const v8bf*)(base + 16);  // e=8..15 : k = k0+16+half*8+(e-8)
  return cat16(lo, hi);
}

// B 32x16 bf16 from a buffer p laid out as p[col][k] (i.e. B transposed,
// contiguous in k): lanes 0-15 hold K {0..7,16..23} of col n=lane, lanes 16-31 K {8..15,24..31}.
__device__ __forceinline__ v16bf frag_bT(const bf16_t* __restrict__ p, int ld,
                                         int k0, int col0, int lane) {
  const int half = (lane >> 4) & 1;
  const int n = lane & 15;
  const bf16_t* base = p + (col0 + n) * ld + k0 + half * 8;
  const v8bf lo = *(const v8bf*)(base);      // e=0..7  : k = half*8+e
  const v8bf hi = *(const v8bf*)(base + 16); // e=8..15 : k = 16+half*8+(e-8)
  return cat16(lo, hi);
}

__device__ __forceinline__ v8f wmma_bf16(v16bf a, v16bf b, v8f c) {
  return __builtin_amdgcn_wmma_f32_16x16x32_bf16(false, a, false, b,
                                                 (short)0, c, false, false);
}

__device__ __forceinline__ void zero8(v8f& v) {
#pragma unroll
  for (int q = 0; q < 8; ++q) v[q] = 0.f;
}

__device__ __forceinline__ float f4c(const float4& f, int q) {
  return q == 0 ? f.x : (q == 1 ? f.y : (q == 2 ? f.z : f.w));
}

// Stage a 256(c) x 128(n) f32 tile of x into LDS transposed: XsT[n][c] bf16.
// Loads stay coalesced (float4 per row); stores packed 4-wide along c (b64).
__device__ __forceinline__ void stage_xT(const float* __restrict__ xb, int n0,
                                         bf16_t* __restrict__ XsT,
                                         int w, int lane) {
  for (int c0 = w * 4; c0 < C_IN; c0 += 32) {
    float4 f[4];
#pragma unroll
    for (int j = 0; j < 4; ++j)
      f[j] = *(const float4*)(xb + (size_t)(c0 + j) * N_TOT + n0 + lane * 4);
#pragma unroll
    for (int q = 0; q < 4; ++q) {
      v4bf pk;
      pk[0] = (bf16_t)f4c(f[0], q);
      pk[1] = (bf16_t)f4c(f[1], q);
      pk[2] = (bf16_t)f4c(f[2], q);
      pk[3] = (bf16_t)f4c(f[3], q);
      *(v4bf*)(XsT + (lane * 4 + q) * C_IN + c0) = pk;
    }
  }
}

// One k-step of the 2-rowtile x 8-coltile GEMM: hoist all 8 B fragments into
// registers first (lets the scheduler clause the ds loads and overlap WMMAs),
// then issue 16 WMMAs.
__device__ __forceinline__ void gemm_step(const bf16_t* __restrict__ Wp, int wld,
                                          const bf16_t* __restrict__ XsT,
                                          int rt0, int ks, int lane,
                                          v8f acc[2][8]) {
  v16bf bb[8];
#pragma unroll
  for (int ct = 0; ct < 8; ++ct) bb[ct] = frag_bT(XsT, C_IN, ks, ct * 16, lane);
  const v16bf a0 = frag_a(Wp, wld, rt0 * 16, ks, lane);
  const v16bf a1 = frag_a(Wp, wld, (rt0 + 1) * 16, ks, lane);
#pragma unroll
  for (int ct = 0; ct < 8; ++ct) {
    acc[0][ct] = wmma_bf16(a0, bb[ct], acc[0][ct]);
    acc[1][ct] = wmma_bf16(a1, bb[ct], acc[1][ct]);
  }
}

// ---- Kernel 0: Wqkv rows 128..383 (k,v weights) -> bf16 ----
__global__ __launch_bounds__(256) void k0_convert(const float* __restrict__ Wqkv,
                                                  bf16_t* __restrict__ Wkv) {
  const int i = blockIdx.x * 256 + threadIdx.x;   // 0..65535
  Wkv[i] = (bf16_t)Wqkv[128 * 256 + i];
}

// ---- Kernel 1: per (b, n-tile): K,V = Wkv @ X;  E=exp(K);  S=rowsum(E);
//      N = E @ V^T (per head).  Deterministic partials per workgroup. ----
__global__ __launch_bounds__(256) void k1_stats(const float* __restrict__ x,
                                                const bf16_t* __restrict__ Wkv,
                                                float* __restrict__ partials) {
  extern __shared__ char smem_raw[];
  bf16_t* XsT = (bf16_t*)smem_raw;          // [128 n][256 c] bf16 (64 KB)
  bf16_t* Es  = XsT + TILE_N * C_IN;        // [128 d][128 j]      (32 KB)
  bf16_t* Vs  = Es + 128 * TILE_N;          // [128 e][128 j]      (32 KB)

  const int tid  = threadIdx.x;
  const int lane = tid & 31;
  const int w    = tid >> 5;               // 8 waves
  const int b    = blockIdx.y;
  const int tile = blockIdx.x;
  const int n0   = tile * TILE_N;
  const float* xb = x + (size_t)b * C_IN * N_TOT;

  stage_xT(xb, n0, XsT, w, lane);
  __syncthreads();

  // KV(256x128) = Wkv(256x256) @ X; wave w owns row-tiles 2w, 2w+1.
  v8f acc[2][8];
#pragma unroll
  for (int i = 0; i < 2; ++i)
#pragma unroll
    for (int j = 0; j < 8; ++j) zero8(acc[i][j]);

  const int rt0 = 2 * w;
#pragma unroll
  for (int ks = 0; ks < C_IN; ks += 32)
    gemm_step(Wkv, C_IN, XsT, rt0, ks, lane, acc);

  // rows 0..127 = K -> exp -> Es ; rows 128..255 = V -> Vs  (row-major [row][j])
  const int half = (lane >> 4) & 1;
  const int ncol = lane & 15;
#pragma unroll
  for (int i = 0; i < 2; ++i) {
    const int rowbase = (rt0 + i) * 16;
#pragma unroll
    for (int ct = 0; ct < 8; ++ct) {
#pragma unroll
      for (int r = 0; r < 8; ++r) {
        const int R = rowbase + r + 8 * half;
        const int j = ct * 16 + ncol;
        const float v = acc[i][ct][r];
        if (R < 128) Es[R * TILE_N + j] = (bf16_t)__expf(v);
        else         Vs[(R - 128) * TILE_N + j] = (bf16_t)v;
      }
    }
  }
  __syncthreads();

  float* pb = partials + (size_t)(b * NTILES + tile) * PART_STRIDE;

  // deterministic per-row sum of E (wide b128 reads, fixed order)
  if (tid < 128) {
    float s = 0.f;
    const bf16_t* row = Es + tid * TILE_N;
#pragma unroll
    for (int jc = 0; jc < TILE_N; jc += 8) {
      const v8bf v = *(const v8bf*)(row + jc);
#pragma unroll
      for (int q = 0; q < 8; ++q) s += (float)v[q];
    }
    pb[tid] = s;
  }

  // N[h][d][e] = sum_j E[h*32+d][j] * V[h*32+e][j] : 16 tiles, 2 per wave
  // A = Es rows (frag_a, ld=128); B = V^T (frag_bT over row-major Vs[e][j])
#pragma unroll
  for (int ti = 0; ti < 2; ++ti) {
    const int t = 2 * w + ti;
    const int h = t >> 2, dt = (t >> 1) & 1, et = t & 1;
    v8f nacc; zero8(nacc);
#pragma unroll
    for (int ks = 0; ks < TILE_N; ks += 32) {
      const v16bf a  = frag_a (Es, TILE_N, h * 32 + dt * 16, ks, lane);
      const v16bf bT = frag_bT(Vs, TILE_N, ks, h * 32 + et * 16, lane);
      nacc = wmma_bf16(a, bT, nacc);
    }
#pragma unroll
    for (int r = 0; r < 8; ++r) {
      const int d = dt * 16 + r + 8 * half;
      const int e = et * 16 + ncol;
      pb[128 + h * 1024 + d * 32 + e] = nacc[r];
    }
  }
}

// ---- Kernel 2a: fixed-order reduction of partials over NTILES ----
__global__ __launch_bounds__(256) void k2a_reduce(const float* __restrict__ partials,
                                                  float* __restrict__ SN) {
  const int gid = blockIdx.x * 256 + threadIdx.x;
  if (gid >= 2 * PART_STRIDE) return;
  const int b = gid / PART_STRIDE;
  const int idx = gid % PART_STRIDE;
  const float* p = partials + (size_t)b * NTILES * PART_STRIDE + idx;
  float s = 0.f;
  for (int t = 0; t < NTILES; ++t) s += p[(size_t)t * PART_STRIDE];
  SN[gid] = s;
}

// ---- Kernel 2b: W3_b[o][c] = sum_hd (sum_e Wout[o][h*32+e]*N[h][d][e]/S[hd])
//                              * scale * Wqkv_q[hd][c]   -> bf16 ----
__global__ __launch_bounds__(128) void k2b_w3(const float* __restrict__ SN,
                                              const float* __restrict__ Wqkv,
                                              const float* __restrict__ Wout,
                                              bf16_t* __restrict__ W3) {
  const int o = blockIdx.x;    // 0..255
  const int b = blockIdx.y;    // 0..1
  const int t = threadIdx.x;   // 0..127 == hd
  __shared__ float wtmp[128];
  const float* S  = SN + (size_t)b * PART_STRIDE;
  const float* Nm = S + 128;
  const int h = t >> 5, d = t & 31;
  float s = 0.f;
  for (int e = 0; e < 32; ++e)
    s += Wout[o * 128 + h * 32 + e] * Nm[h * 1024 + d * 32 + e];
  wtmp[t] = s / S[t];
  __syncthreads();
  const float scale = 0.17677669529663687f;   // 32^-0.5
  for (int c = t; c < 256; c += 128) {
    float a = 0.f;
    for (int hd = 0; hd < 128; ++hd) a += wtmp[hd] * Wqkv[hd * 256 + c];
    W3[((size_t)b * 256 + o) * 256 + c] = (bf16_t)(a * scale);
  }
}

// ---- Kernel 3: y[b] = W3_b(256x256) @ x[b] + b_out ----
__global__ __launch_bounds__(256) void k3_out(const float* __restrict__ x,
                                              const bf16_t* __restrict__ W3,
                                              const float* __restrict__ bout,
                                              float* __restrict__ y) {
  extern __shared__ char smem_raw[];
  bf16_t* XsT = (bf16_t*)smem_raw;  // [128 n][256 c] bf16
  const int tid = threadIdx.x, lane = tid & 31, w = tid >> 5;
  const int b = blockIdx.y, n0 = blockIdx.x * TILE_N;
  const float* xb = x + (size_t)b * C_IN * N_TOT;
  const bf16_t* Wb = W3 + (size_t)b * 256 * 256;

  stage_xT(xb, n0, XsT, w, lane);
  __syncthreads();

  v8f acc[2][8];
#pragma unroll
  for (int i = 0; i < 2; ++i)
#pragma unroll
    for (int j = 0; j < 8; ++j) zero8(acc[i][j]);

  const int rt0 = 2 * w;
#pragma unroll
  for (int ks = 0; ks < C_IN; ks += 32)
    gemm_step(Wb, C_IN, XsT, rt0, ks, lane, acc);

  const int half = (lane >> 4) & 1, n = lane & 15;
  float* yb = y + (size_t)b * C_IN * N_TOT;
#pragma unroll
  for (int i = 0; i < 2; ++i) {
#pragma unroll
    for (int ct = 0; ct < 8; ++ct) {
#pragma unroll
      for (int r = 0; r < 8; ++r) {
        const int R = (rt0 + i) * 16 + r + 8 * half;
        const int col = n0 + ct * 16 + n;
        yb[(size_t)R * N_TOT + col] = acc[i][ct][r] + bout[R];
      }
    }
  }
}

extern "C" void kernel_launch(void* const* d_in, const int* in_sizes, int n_in,
                              void* d_out, int out_size, void* d_ws, size_t ws_size,
                              hipStream_t stream) {
  const float* x    = (const float*)d_in[0];
  const float* Wqkv = (const float*)d_in[1];
  const float* Wout = (const float*)d_in[2];
  const float* bout = (const float*)d_in[3];
  float* y = (float*)d_out;

  char* ws = (char*)d_ws;
  bf16_t* Wkv   = (bf16_t*)(ws);                    // 131072 B
  bf16_t* W3    = (bf16_t*)(ws + 131072);           // 262144 B
  float*  SN    = (float*) (ws + 131072 + 262144);  // 2*4224*4 = 33792 B
  float*  parts = (float*) (ws + 458752);           // 1024*4224*4 ≈ 17.3 MB

  k0_convert<<<dim3(256),       dim3(256), 0,      stream>>>(Wqkv, Wkv);
  k1_stats  <<<dim3(NTILES, 2), dim3(256), 131072, stream>>>(x, Wkv, parts);
  k2a_reduce<<<dim3(33),        dim3(256), 0,      stream>>>(parts, SN);
  k2b_w3    <<<dim3(256, 2),    dim3(128), 0,      stream>>>(SN, Wqkv, Wout, W3);
  k3_out    <<<dim3(NTILES, 2), dim3(256), 65536,  stream>>>(x, W3, bout, y);
}
